// MultivariateCausalFlow_31576599560778
// MI455X (gfx1250) — matrix operand order repaired and checked by reading, not compiled, not run on hardware
//
#include <hip/hip_runtime.h>
#include <hip/hip_bf16.h>
#include <stdint.h>

#define DIMN  64
#define KD    32
#define NHID  256
#define TOT   2048
#define BATCH 4096

typedef __attribute__((ext_vector_type(16))) __bf16 v16bf;
typedef __attribute__((ext_vector_type(8)))  float  v8f;

struct Frag { uint4 q0, q1; };   // 32 bytes = 16 bf16 = one WMMA A/B operand

static __device__ __forceinline__ unsigned short f2bf(float x) {
    unsigned u = __builtin_bit_cast(unsigned, x);
    unsigned r = u + 0x7FFFu + ((u >> 16) & 1u);   // round-to-nearest-even
    return (unsigned short)(r >> 16);
}

// ---------------------------------------------------------------------------
// Pack an f32 row-major [Krows x Ncols] weight matrix into bf16 fragment-major
// layout matching the CDNA5 16-bit B-matrix VGPR layout:
//   fragment (kblk, ntile): VGPR p, lane h*16+n = { W[kblk*32+8h+2p+(p>=4?8:0), ntile*16+n],
//                                                   W[kblk*32+8h+2p+(p>=4?8:0)+1, ...] }
// stored as dst[(frag*32+lane)*8 + p] so a lane's 8 dwords are contiguous.
// ---------------------------------------------------------------------------
__global__ void pack_weight_frags(const float* __restrict__ src,
                                  uint32_t* __restrict__ dst,
                                  int Krows, int Ncols) {
    int tid = blockIdx.x * blockDim.x + threadIdx.x;
    int Ntiles = Ncols >> 4;
    int totalFrags = (Krows >> 5) * Ntiles;
    int frag = tid >> 5;
    if (frag >= totalFrags) return;
    int lane = tid & 31;
    int h = lane >> 4, n = lane & 15;
    int kblk = frag / Ntiles, ntile = frag % Ntiles;
#pragma unroll
    for (int p = 0; p < 8; ++p) {
        int k0 = kblk * 32 + 8 * h + 2 * p + (p >= 4 ? 8 : 0);
        unsigned lo = f2bf(src[(size_t)k0 * Ncols + ntile * 16 + n]);
        unsigned hi = f2bf(src[(size_t)(k0 + 1) * Ncols + ntile * 16 + n]);
        dst[(size_t)tid * 8 + p] = lo | (hi << 16);
    }
}

// NT consecutive B fragments starting at fragBase, processed in groups of G
// (all G fragments simultaneously live -> distinct registers -> one clause of
// 2*G b128 loads, then a 2*G WMMA burst; each B feeds both 16-row A strips).
template <int NT, int G>
static __device__ __forceinline__ void gemm_tiles2(const uint32_t* __restrict__ W,
                                                   int fragBase, int lane,
                                                   v16bf a0, v16bf a1,
                                                   v8f* acc0, v8f* acc1) {
    const uint4* wp = (const uint4*)(W + ((size_t)fragBase * 32 + lane) * 8);
    // fragment stride = 32 lanes * 8 dwords = 64 uint4
#pragma unroll
    for (int g = 0; g < NT; g += G) {
        Frag b[G];
#pragma unroll
        for (int u = 0; u < G; ++u) {
            b[u].q0 = wp[(g + u) * 64];
            b[u].q1 = wp[(g + u) * 64 + 1];
        }
#pragma unroll
        for (int u = 0; u < G; ++u) {
            v16bf bv = __builtin_bit_cast(v16bf, b[u]);
            acc0[g + u] = __builtin_amdgcn_wmma_f32_16x16x32_bf16(
                false, a0, false, bv, (short)0, acc0[g + u], false, false);
            acc1[g + u] = __builtin_amdgcn_wmma_f32_16x16x32_bf16(
                false, a1, false, bv, (short)0, acc1[g + u], false, false);
        }
    }
}

// ---------------------------------------------------------------------------
// One causal-flow step (step i). Grid: BATCH/32 blocks x 256 threads (8 waves).
// Wave = (mlp = wave>>2 : 0=s-net 1=t-net) x (nq = wave&3 : 64-wide N quarter).
// Each wave: M = 32 batch rows (both strips), N = 4 tiles -> 64 acc VGPRs.
// ---------------------------------------------------------------------------
__global__ __launch_bounds__(256)
void flow_step(int i,
               float* __restrict__ z_out, float* __restrict__ ld_out,
               const float* __restrict__ e, const int* __restrict__ C,
               const uint32_t* __restrict__ W1s, const uint32_t* __restrict__ W1t,
               const uint32_t* __restrict__ W2s, const uint32_t* __restrict__ W2t,
               const uint32_t* __restrict__ W3s, const uint32_t* __restrict__ W3t,
               const float* __restrict__ b1s, const float* __restrict__ b2s,
               const float* __restrict__ b3s,
               const float* __restrict__ b1t, const float* __restrict__ b2t,
               const float* __restrict__ b3t,
               unsigned short* __restrict__ zbf, float* __restrict__ ld_ws) {
    extern __shared__ char smem[];
    const int HS = 264;                       // padded bf16 row stride (bank-conflict free)
    unsigned short* h1 = (unsigned short*)smem;                 // [2][32][264] bf16
    unsigned short* h2 = (unsigned short*)(smem + 2 * 32 * HS * 2);
    float* out3 = (float*)(smem + 2 * 2 * 32 * HS * 2);         // [2][32][32] f32
    int*   plist = (int*)(smem + 75776);                        // [64] + count
    int*   pcnt  = plist + 64;
    float* psum  = (float*)(smem + 76048);                      // [256] (also flags[64] early)
    int*   flags = (int*)psum;

    const int tid = threadIdx.x;
    const int wave = tid >> 5;
    const int lane = tid & 31;
    const int h = lane >> 4;
    const int n = lane & 15;
    const int mlp = wave >> 2;     // 0 = s-net, 1 = t-net
    const int nq  = wave & 3;      // which 64-wide quarter of the hidden layer
    const int rowBase = blockIdx.x * 32;

    const uint32_t* W1 = mlp ? W1t : W1s;
    const uint32_t* W2 = mlp ? W2t : W2s;
    const uint32_t* W3 = mlp ? W3t : W3s;
    const float* b1 = mlp ? b1t : b1s;
    const float* b2 = mlp ? b2t : b2s;
    const float* b3 = mlp ? b3t : b3s;

    // ---------------- build active-parent list (parallel flags + compaction)
    if (tid < DIMN) flags[tid] = (tid < i && C[tid * DIMN + i] != 0) ? 1 : 0;
    __syncthreads();
    if (tid == 0) {
        int c = 0;
        for (int j = 0; j < DIMN; ++j)
            if (flags[j]) plist[c++] = j;
        *pcnt = c;
    }
    __syncthreads();
    const int cnt = *pcnt;

    // ---------------- Layer 1: h1 = relu(ctx @ W1 + b1), K = active parents
    // acc[0..3] = strip 0 (rows 0-15), acc[4..7] = strip 1 (rows 16-31)
    v8f acc[8];
#pragma unroll
    for (int t = 0; t < 4; ++t) {
        float bv = b1[(nq * 4 + t) * 16 + n];
        v8f a = {bv, bv, bv, bv, bv, bv, bv, bv};
        acc[t] = a;
        acc[4 + t] = a;
    }
    Frag a0c, a1c, a0n, a1n;
    if (cnt > 0) {
        const unsigned short* z0 = zbf + (size_t)(rowBase + n) * TOT + plist[0] * 32 + h * 8;
        const unsigned short* z1 = z0 + (size_t)16 * TOT;
        a0c.q0 = *(const uint4*)(z0); a0c.q1 = *(const uint4*)(z0 + 16);
        a1c.q0 = *(const uint4*)(z1); a1c.q1 = *(const uint4*)(z1 + 16);
    }
    for (int idx = 0; idx < cnt; ++idx) {
        int j = plist[idx];
        if (idx + 1 < cnt) {   // prefetch next parent's A fragments (both strips)
            const unsigned short* z0 = zbf + (size_t)(rowBase + n) * TOT + plist[idx + 1] * 32 + h * 8;
            const unsigned short* z1 = z0 + (size_t)16 * TOT;
            a0n.q0 = *(const uint4*)(z0); a0n.q1 = *(const uint4*)(z0 + 16);
            a1n.q0 = *(const uint4*)(z1); a1n.q1 = *(const uint4*)(z1 + 16);
        }
        gemm_tiles2<4, 4>(W1, j * 16 + nq * 4, lane,
                          __builtin_bit_cast(v16bf, a0c), __builtin_bit_cast(v16bf, a1c),
                          acc, acc + 4);
        a0c = a0n; a1c = a1n;
    }
#pragma unroll
    for (int t = 0; t < 4; ++t)
#pragma unroll
        for (int v = 0; v < 8; ++v) {
            int col = (nq * 4 + t) * 16 + n;
            h1[(mlp * 32 + (v + 8 * h)) * HS + col]      = f2bf(fmaxf(acc[t][v], 0.0f));
            h1[(mlp * 32 + (16 + v + 8 * h)) * HS + col] = f2bf(fmaxf(acc[4 + t][v], 0.0f));
        }
    __syncthreads();   // layer-2 A fragments read ALL column quarters of h1

    // ---------------- Layer 2: h2 = relu(h1 @ W2 + b2), K = 256
#pragma unroll
    for (int t = 0; t < 4; ++t) {
        float bv = b2[(nq * 4 + t) * 16 + n];
        v8f a = {bv, bv, bv, bv, bv, bv, bv, bv};
        acc[t] = a;
        acc[4 + t] = a;
    }
    for (int kb = 0; kb < 8; ++kb) {
        const unsigned short* p0 = h1 + (size_t)(mlp * 32 + n) * HS + kb * 32 + h * 8;
        const unsigned short* p1 = p0 + (size_t)16 * HS;
        Frag a0, a1;
        a0.q0 = *(const uint4*)(p0); a0.q1 = *(const uint4*)(p0 + 16);
        a1.q0 = *(const uint4*)(p1); a1.q1 = *(const uint4*)(p1 + 16);
        gemm_tiles2<4, 4>(W2, kb * 16 + nq * 4, lane,
                          __builtin_bit_cast(v16bf, a0), __builtin_bit_cast(v16bf, a1),
                          acc, acc + 4);
    }
#pragma unroll
    for (int t = 0; t < 4; ++t)
#pragma unroll
        for (int v = 0; v < 8; ++v) {
            int col = (nq * 4 + t) * 16 + n;
            h2[(mlp * 32 + (v + 8 * h)) * HS + col]      = f2bf(fmaxf(acc[t][v], 0.0f));
            h2[(mlp * 32 + (16 + v + 8 * h)) * HS + col] = f2bf(fmaxf(acc[4 + t][v], 0.0f));
        }
    __syncthreads();   // layer-3 A fragments read ALL column quarters of h2

    // ---------------- Layer 3: out = h2 @ W3 + b3, N = 32 -> waves nq<2 only
    if (nq < 2) {
        v8f acc3[2];
        {
            float bv = b3[nq * 16 + n];
            v8f a = {bv, bv, bv, bv, bv, bv, bv, bv};
            acc3[0] = a;
            acc3[1] = a;
        }
        for (int kb = 0; kb < 8; ++kb) {
            const unsigned short* p0 = h2 + (size_t)(mlp * 32 + n) * HS + kb * 32 + h * 8;
            const unsigned short* p1 = p0 + (size_t)16 * HS;
            Frag a0, a1;
            a0.q0 = *(const uint4*)(p0); a0.q1 = *(const uint4*)(p0 + 16);
            a1.q0 = *(const uint4*)(p1); a1.q1 = *(const uint4*)(p1 + 16);
            gemm_tiles2<1, 1>(W3, kb * 2 + nq, lane,
                              __builtin_bit_cast(v16bf, a0), __builtin_bit_cast(v16bf, a1),
                              acc3, acc3 + 1);
        }
#pragma unroll
        for (int v = 0; v < 8; ++v) {
            int col = nq * 16 + n;
            out3[(mlp * 32 + (v + 8 * h)) * 32 + col]      = acc3[0][v];
            out3[(mlp * 32 + (16 + v + 8 * h)) * 32 + col] = acc3[1][v];
        }
    }

    __syncthreads();

    // ---------------- Epilogue: z_i = exp(s)*e_i + t ; log_det += sum(s)
    {
        int r = tid >> 3;        // 0..31 : block-local batch row
        int q = tid & 7;         // 0..7  : 4-column chunk
        size_t grow = (size_t)rowBase + r;
        float partial = 0.0f;
#pragma unroll
        for (int cc = 0; cc < 4; ++cc) {
            int c = q * 4 + cc;
            float sraw = out3[r * 32 + c];
            float traw = out3[(32 + r) * 32 + c];
            float s = 0.1f * tanhf(sraw);
            size_t idx = grow * TOT + (size_t)i * 32 + c;
            float zi = expf(s) * e[idx] + traw;
            z_out[idx] = zi;
            zbf[idx] = f2bf(zi);
            partial += s;
        }
        psum[tid] = partial;
    }
    __syncthreads();
    if (tid < 32) {
        size_t grow = (size_t)rowBase + tid;
        float sum = 0.0f;
#pragma unroll
        for (int q = 0; q < 8; ++q) sum += psum[tid * 8 + q];
        float ld = (i == 0) ? sum : (ld_ws[grow] + sum);
        ld_ws[grow] = ld;
        if (i == DIMN - 1) ld_out[grow] = ld;
    }
}

// ---------------------------------------------------------------------------
extern "C" void kernel_launch(void* const* d_in, const int* in_sizes, int n_in,
                              void* d_out, int out_size, void* d_ws, size_t ws_size,
                              hipStream_t stream) {
    (void)in_sizes; (void)n_in; (void)out_size; (void)ws_size;

    const float* e   = (const float*)d_in[0];
    const int*   C   = (const int*)d_in[1];
    const float* sW1 = (const float*)d_in[2];
    const float* sb1 = (const float*)d_in[3];
    const float* sW2 = (const float*)d_in[4];
    const float* sb2 = (const float*)d_in[5];
    const float* sW3 = (const float*)d_in[6];
    const float* sb3 = (const float*)d_in[7];
    const float* tW1 = (const float*)d_in[8];
    const float* tb1 = (const float*)d_in[9];
    const float* tW2 = (const float*)d_in[10];
    const float* tb2 = (const float*)d_in[11];
    const float* tW3 = (const float*)d_in[12];
    const float* tb3 = (const float*)d_in[13];

    float* z_out  = (float*)d_out;
    float* ld_out = z_out + (size_t)BATCH * TOT;

    // workspace layout
    char* ws = (char*)d_ws;
    size_t off = 0;
    unsigned short* zbf = (unsigned short*)(ws + off); off += (size_t)BATCH * TOT * 2;   // 16 MB
    uint32_t* W1s = (uint32_t*)(ws + off); off += (size_t)TOT * NHID * 2;
    uint32_t* W1t = (uint32_t*)(ws + off); off += (size_t)TOT * NHID * 2;
    uint32_t* W2s = (uint32_t*)(ws + off); off += (size_t)NHID * NHID * 2;
    uint32_t* W2t = (uint32_t*)(ws + off); off += (size_t)NHID * NHID * 2;
    uint32_t* W3s = (uint32_t*)(ws + off); off += (size_t)NHID * KD * 2;
    uint32_t* W3t = (uint32_t*)(ws + off); off += (size_t)NHID * KD * 2;
    float* ld_ws  = (float*)(ws + off);    off += (size_t)BATCH * 4;

    // pack all six weight matrices into WMMA fragment-major bf16
    auto pack = [&](const float* src, uint32_t* dst, int K, int N) {
        int threads = (K / 32) * (N / 16) * 32;
        pack_weight_frags<<<(threads + 255) / 256, 256, 0, stream>>>(src, dst, K, N);
    };
    pack(sW1, W1s, TOT,  NHID);
    pack(tW1, W1t, TOT,  NHID);
    pack(sW2, W2s, NHID, NHID);
    pack(tW2, W2t, NHID, NHID);
    pack(sW3, W3s, NHID, KD);
    pack(tW3, W3t, NHID, KD);

    // LDS: h1(33792) + h2(33792) + out3(8192) + plist/count(272) + psum(1024)
    const int ldsBytes = 77072;
    hipFuncSetAttribute((const void*)flow_step,
                        hipFuncAttributeMaxDynamicSharedMemorySize, ldsBytes);

    for (int i = 0; i < DIMN; ++i) {
        flow_step<<<BATCH / 32, 256, ldsBytes, stream>>>(
            i, z_out, ld_out, e, C,
            W1s, W1t, W2s, W2t, W3s, W3t,
            sb1, sb2, sb3, tb1, tb2, tb3,
            zbf, ld_ws);
    }
}
